// EncoderLayer_17798344475169
// MI455X (gfx1250) — matrix operand order repaired and checked by reading, not compiled, not run on hardware
//
#include <hip/hip_runtime.h>
#include <hip/hip_bf16.h>

// EncoderLayer for MI455X (gfx1250): wave32 WMMA f16->f32 throughout.
// B=8, T=2048, C=64, H=6, HS=64, FF=256.

#define B_  8
#define T_  2048
#define C_  64
#define H_  6
#define HS_ 64
#define FF_ 256

typedef _Float16 h8  __attribute__((ext_vector_type(8)));
typedef _Float16 h16 __attribute__((ext_vector_type(16)));
typedef float    f8  __attribute__((ext_vector_type(8)));

__device__ __forceinline__ f8 wmma16(h16 a, h16 b, f8 c) {
  // D = A(16x32 f16) * B(32x16 f16) + C(16x16 f32)
  return __builtin_amdgcn_wmma_f32_16x16x32_f16(false, a, false, b, (short)0, c,
                                                false, false);
}

// A-operand (16x32 f16) loader for one lane.
// Lane L holds row m = L%16. hi = L>>4.
// halves 0..7  -> k = col0 + 16*hi + i
// halves 8..15 -> k = col0 + 16*hi + 8 + i
__device__ __forceinline__ h16 load_a16(const _Float16* row, int hi, int col0) {
  h8 lo = *(const h8*)(row + col0 + 16 * hi);
  h8 hv = *(const h8*)(row + col0 + 16 * hi + 8);
  h16 a;
#pragma unroll
  for (int i = 0; i < 8; ++i) { a[i] = lo[i]; a[i + 8] = hv[i]; }
  return a;
}

// ---------------------------------------------------------------------------
// Conversion kernels: fp32 -> f16, weights transposed so that the WMMA
// B-operand (lane n = L%16, k = h + 16*hi) is a contiguous 16-half load.
// ---------------------------------------------------------------------------
__global__ void conv_x_kernel(const float* __restrict__ x,
                              _Float16* __restrict__ xh) {
  int i = blockIdx.x * blockDim.x + threadIdx.x;
  if (i < B_ * T_ * C_) xh[i] = (_Float16)x[i];
}

__global__ void conv_w_kernel(const float* __restrict__ Wq,
                              const float* __restrict__ Wk,
                              const float* __restrict__ Wv,
                              const float* __restrict__ Wp,
                              const float* __restrict__ W1,
                              const float* __restrict__ W2,
                              _Float16* __restrict__ wqT,
                              _Float16* __restrict__ wkT,
                              _Float16* __restrict__ wvT,
                              _Float16* __restrict__ wpT,
                              _Float16* __restrict__ w1T,
                              _Float16* __restrict__ w2T) {
  int i = blockIdx.x * blockDim.x + threadIdx.x;
  if (i < 73728) {                     // WqT/WkT/WvT: [H][HS][C] <- [H][C][HS]
    int w = i / 24576, j = i % 24576;
    int h = j / 4096, rem = j % 4096, d = rem / 64, c = rem % 64;
    const float* src = (w == 0) ? Wq : (w == 1) ? Wk : Wv;
    _Float16* dst = (w == 0) ? wqT : (w == 1) ? wkT : wvT;
    dst[j] = (_Float16)src[h * 4096 + c * 64 + d];
  } else if (i < 98304) {              // WpT: [64][384] <- [384][64]
    int j = i - 73728, n = j / 384, k = j % 384;
    wpT[j] = (_Float16)Wp[k * 64 + n];
  } else if (i < 114688) {             // W1T: [256][64] <- [64][256]
    int j = i - 98304, n = j / 64, k = j % 64;
    w1T[j] = (_Float16)W1[k * 256 + n];
  } else if (i < 131072) {             // W2T: [64][256] <- [256][64]
    int j = i - 114688, n = j / 256, k = j % 256;
    w2T[j] = (_Float16)W2[k * 64 + n];
  }
}

// ---------------------------------------------------------------------------
// QKV projection: per (b, h, 16-row tile): q/k/v = x(16x64) @ W(64x64).
// q scaled by 1/sqrt(HS). v stored transposed [d][t] for the PV B-operand.
// ---------------------------------------------------------------------------
__global__ void qkv_kernel(const _Float16* __restrict__ xh,
                           const _Float16* __restrict__ wqT,
                           const _Float16* __restrict__ wkT,
                           const _Float16* __restrict__ wvT,
                           _Float16* __restrict__ q,
                           _Float16* __restrict__ k,
                           _Float16* __restrict__ vT) {
  const int lane = threadIdx.x & 31;
  const int n = lane & 15;
  const int hi = lane >> 4;
  int bx = blockIdx.x;
  const int tile = bx % (T_ / 16);
  const int h = (bx / (T_ / 16)) % H_;
  const int b = bx / ((T_ / 16) * H_);
  const int t0 = tile * 16;

  const _Float16* xrow = xh + (size_t)(b * T_ + t0 + n) * C_;  // A row, m=n
  h16 a0 = load_a16(xrow, hi, 0);
  h16 a1 = load_a16(xrow, hi, 32);

  const _Float16* wT[3] = {wqT + h * 4096, wkT + h * 4096, wvT + h * 4096};
  const size_t hb = (size_t)(b * H_ + h);

#pragma unroll
  for (int w = 0; w < 3; ++w) {
#pragma unroll
    for (int nt = 0; nt < 4; ++nt) {
      f8 acc;
#pragma unroll
      for (int i = 0; i < 8; ++i) acc[i] = 0.0f;
      const _Float16* brow = wT[w] + (nt * 16 + n) * 64;   // row n of W^T
      acc = wmma16(a0, *(const h16*)(brow + 16 * hi), acc);
      acc = wmma16(a1, *(const h16*)(brow + 32 + 16 * hi), acc);
#pragma unroll
      for (int r = 0; r < 8; ++r) {
        int m = r + 8 * hi;
        int t = t0 + m;
        int d = nt * 16 + n;
        float val = acc[r];
        if (w == 0)       q[(hb * T_ + t) * HS_ + d] = (_Float16)(val * 0.125f);
        else if (w == 1)  k[(hb * T_ + t) * HS_ + d] = (_Float16)val;
        else              vT[(hb * HS_ + d) * T_ + t] = (_Float16)val;
      }
    }
  }
}

// ---------------------------------------------------------------------------
// Flash attention: one wave per (b, h, 16-query tile); streams 32 keys/iter.
// Online softmax in fp32; P re-shaped C/D->A via LDS; O accumulated via WMMA.
// Writes o as f16 in [B*T][H*HS] (heads concatenated) for the projection.
// ---------------------------------------------------------------------------
__global__ void attn_kernel(const _Float16* __restrict__ q,
                            const _Float16* __restrict__ k,
                            const _Float16* __restrict__ vT,
                            _Float16* __restrict__ o) {
  __shared__ _Float16 sP[16][40];  // 16x32 probs, padded row = 80B (16B mult.)

  const int lane = threadIdx.x & 31;
  const int n = lane & 15;
  const int hi = lane >> 4;
  int bx = blockIdx.x;
  const int tile = bx % (T_ / 16);
  const int h = (bx / (T_ / 16)) % H_;
  const int b = bx / ((T_ / 16) * H_);
  const int t0 = tile * 16;
  const size_t hb = (size_t)(b * H_ + h);

  const _Float16* qrow = q + (hb * T_ + t0 + n) * HS_;
  h16 qa0 = load_a16(qrow, hi, 0);
  h16 qa1 = load_a16(qrow, hi, 32);

  const _Float16* kbase = k + hb * T_ * HS_;
  const _Float16* vbase = vT + hb * HS_ * T_;

  f8 oacc[4];
#pragma unroll
  for (int nt = 0; nt < 4; ++nt)
#pragma unroll
    for (int i = 0; i < 8; ++i) oacc[nt][i] = 0.0f;

  float mrow[8], lrow[8];
#pragma unroll
  for (int r = 0; r < 8; ++r) { mrow[r] = -3.0e38f; lrow[r] = 0.0f; }

  for (int kt = 0; kt < T_ / 32; ++kt) {
    // speculative prefetch of next K/V tiles (global_prefetch_b8)
    __builtin_prefetch(kbase + ((size_t)(kt + 1) * 32 + n) * HS_, 0, 1);
    __builtin_prefetch(vbase + (size_t)n * T_ + (kt + 1) * 32, 0, 1);

    // S = Q (16x64) x K^T -> two 16x16 tiles (keys kt*32 .. kt*32+31)
    f8 s[2];
#pragma unroll
    for (int st = 0; st < 2; ++st) {
      f8 acc;
#pragma unroll
      for (int i = 0; i < 8; ++i) acc[i] = 0.0f;
      const _Float16* krow = kbase + (size_t)(kt * 32 + st * 16 + n) * HS_;
      acc = wmma16(qa0, *(const h16*)(krow + 16 * hi), acc);
      acc = wmma16(qa1, *(const h16*)(krow + 32 + 16 * hi), acc);
      s[st] = acc;
    }

    // Online softmax per row (row = r + 8*hi; 16-lane butterfly over n)
    float alpha[8];
#pragma unroll
    for (int r = 0; r < 8; ++r) {
      float mx = fmaxf(s[0][r], s[1][r]);
      mx = fmaxf(mx, __shfl_xor(mx, 1));
      mx = fmaxf(mx, __shfl_xor(mx, 2));
      mx = fmaxf(mx, __shfl_xor(mx, 4));
      mx = fmaxf(mx, __shfl_xor(mx, 8));
      float mnew = fmaxf(mrow[r], mx);
      float p0 = __expf(s[0][r] - mnew);
      float p1 = __expf(s[1][r] - mnew);
      float ps = p0 + p1;
      ps += __shfl_xor(ps, 1);
      ps += __shfl_xor(ps, 2);
      ps += __shfl_xor(ps, 4);
      ps += __shfl_xor(ps, 8);
      float a = __expf(mrow[r] - mnew);
      lrow[r] = lrow[r] * a + ps;
      mrow[r] = mnew;
      alpha[r] = a;
      int m = r + 8 * hi;
      sP[m][n]      = (_Float16)p0;
      sP[m][16 + n] = (_Float16)p1;
    }

    // Rescale running O by alpha (same (r,hi) row mapping as stats)
#pragma unroll
    for (int nt = 0; nt < 4; ++nt)
#pragma unroll
      for (int r = 0; r < 8; ++r) oacc[nt][r] *= alpha[r];

    __syncthreads();

    // P as A-operand (16x32 f16) from LDS; V chunk as B-operand (contig in t)
    h16 pa = load_a16(&sP[n][0], hi, 0);
#pragma unroll
    for (int nt = 0; nt < 4; ++nt) {
      const _Float16* vrow = vbase + (size_t)(nt * 16 + n) * T_ + kt * 32 + 16 * hi;
      oacc[nt] = wmma16(pa, *(const h16*)vrow, oacc[nt]);
    }
    __syncthreads();
  }

  // Finalize: divide by l, store f16 to [B*T][H*HS]
  float inv[8];
#pragma unroll
  for (int r = 0; r < 8; ++r) inv[r] = 1.0f / lrow[r];
#pragma unroll
  for (int nt = 0; nt < 4; ++nt)
#pragma unroll
    for (int r = 0; r < 8; ++r) {
      int m = r + 8 * hi;
      int t = t0 + m;
      int d = nt * 16 + n;
      o[((size_t)(b * T_ + t)) * (H_ * HS_) + h * HS_ + d] =
          (_Float16)(oacc[nt][r] * inv[r]);
    }
}

// ---------------------------------------------------------------------------
// attn-proj + residual + LN1 + FFN(relu) + residual(x) + LN2, fused per tile.
// ---------------------------------------------------------------------------
__global__ void mlp_kernel(const _Float16* __restrict__ o,
                           const _Float16* __restrict__ wpT,
                           const float* __restrict__ x,
                           const float* __restrict__ bp,
                           const float* __restrict__ g1,
                           const float* __restrict__ b1,
                           const _Float16* __restrict__ w1T,
                           const float* __restrict__ bf1,
                           const _Float16* __restrict__ w2T,
                           const float* __restrict__ bf2,
                           const float* __restrict__ g2,
                           const float* __restrict__ b2,
                           float* __restrict__ out) {
  __shared__ _Float16 sY[16][72];   // 16x64 y (f16), row = 144B
  __shared__ _Float16 sH[16][264];  // 16x256 relu(h1) (f16), row = 528B

  const int lane = threadIdx.x & 31;
  const int n = lane & 15;
  const int hi = lane >> 4;
  const int row0 = blockIdx.x * 16;

  // attn = o(16x384) @ Wp(384x64)
  f8 acc[4];
#pragma unroll
  for (int nt = 0; nt < 4; ++nt)
#pragma unroll
    for (int i = 0; i < 8; ++i) acc[nt][i] = 0.0f;

  const _Float16* orow = o + (size_t)(row0 + n) * (H_ * HS_);
#pragma unroll
  for (int c = 0; c < 12; ++c) {
    h16 a = load_a16(orow, hi, c * 32);
#pragma unroll
    for (int nt = 0; nt < 4; ++nt) {
      const _Float16* brow = wpT + (size_t)(nt * 16 + n) * 384 + c * 32 + 16 * hi;
      acc[nt] = wmma16(a, *(const h16*)brow, acc[nt]);
    }
  }

  // + bp + x ; LayerNorm1 -> y (f16 in LDS)
#pragma unroll
  for (int r = 0; r < 8; ++r) {
    int m = r + 8 * hi;
    float sm = 0.0f, sq = 0.0f;
#pragma unroll
    for (int nt = 0; nt < 4; ++nt) {
      int col = nt * 16 + n;
      float v = acc[nt][r] + bp[col] + x[(size_t)(row0 + m) * C_ + col];
      acc[nt][r] = v;
      sm += v; sq += v * v;
    }
    sm += __shfl_xor(sm, 1); sq += __shfl_xor(sq, 1);
    sm += __shfl_xor(sm, 2); sq += __shfl_xor(sq, 2);
    sm += __shfl_xor(sm, 4); sq += __shfl_xor(sq, 4);
    sm += __shfl_xor(sm, 8); sq += __shfl_xor(sq, 8);
    float mu = sm * (1.0f / 64.0f);
    float var = sq * (1.0f / 64.0f) - mu * mu;
    float is = rsqrtf(var + 1e-5f);
#pragma unroll
    for (int nt = 0; nt < 4; ++nt) {
      int col = nt * 16 + n;
      float yv = g1[col] * (acc[nt][r] - mu) * is + b1[col];
      sY[m][col] = (_Float16)yv;
    }
  }
  __syncthreads();

  // h1 = relu(y @ W1 + bf1) -> LDS (one 16-col tile at a time)
  h16 ya0 = load_a16(&sY[n][0], hi, 0);
  h16 ya1 = load_a16(&sY[n][0], hi, 32);
#pragma unroll
  for (int nt = 0; nt < 16; ++nt) {
    f8 hacc;
#pragma unroll
    for (int i = 0; i < 8; ++i) hacc[i] = 0.0f;
    const _Float16* brow = w1T + (size_t)(nt * 16 + n) * 64;
    hacc = wmma16(ya0, *(const h16*)(brow + 16 * hi), hacc);
    hacc = wmma16(ya1, *(const h16*)(brow + 32 + 16 * hi), hacc);
#pragma unroll
    for (int r = 0; r < 8; ++r) {
      int m = r + 8 * hi;
      int col = nt * 16 + n;
      float v = fmaxf(hacc[r] + bf1[col], 0.0f);
      sH[m][col] = (_Float16)v;
    }
  }
  __syncthreads();

  // f = h1(16x256) @ W2(256x64)
  f8 facc[4];
#pragma unroll
  for (int nt = 0; nt < 4; ++nt)
#pragma unroll
    for (int i = 0; i < 8; ++i) facc[nt][i] = 0.0f;
#pragma unroll
  for (int c = 0; c < 8; ++c) {
    h16 ha = load_a16(&sH[n][0], hi, c * 32);
#pragma unroll
    for (int nt = 0; nt < 4; ++nt) {
      const _Float16* brow = w2T + (size_t)(nt * 16 + n) * 256 + c * 32 + 16 * hi;
      facc[nt] = wmma16(ha, *(const h16*)brow, facc[nt]);
    }
  }

  // + bf2 + x (ORIGINAL input, per source) ; LayerNorm2 -> out (fp32)
#pragma unroll
  for (int r = 0; r < 8; ++r) {
    int m = r + 8 * hi;
    float sm = 0.0f, sq = 0.0f;
#pragma unroll
    for (int nt = 0; nt < 4; ++nt) {
      int col = nt * 16 + n;
      float v = facc[nt][r] + bf2[col] + x[(size_t)(row0 + m) * C_ + col];
      facc[nt][r] = v;
      sm += v; sq += v * v;
    }
    sm += __shfl_xor(sm, 1); sq += __shfl_xor(sq, 1);
    sm += __shfl_xor(sm, 2); sq += __shfl_xor(sq, 2);
    sm += __shfl_xor(sm, 4); sq += __shfl_xor(sq, 4);
    sm += __shfl_xor(sm, 8); sq += __shfl_xor(sq, 8);
    float mu = sm * (1.0f / 64.0f);
    float var = sq * (1.0f / 64.0f) - mu * mu;
    float is = rsqrtf(var + 1e-5f);
#pragma unroll
    for (int nt = 0; nt < 4; ++nt) {
      int col = nt * 16 + n;
      out[(size_t)(row0 + m) * C_ + col] =
          g2[col] * (facc[nt][r] - mu) * is + b2[col];
    }
  }
}

// ---------------------------------------------------------------------------
extern "C" void kernel_launch(void* const* d_in, const int* in_sizes, int n_in,
                              void* d_out, int out_size, void* d_ws, size_t ws_size,
                              hipStream_t stream) {
  (void)in_sizes; (void)n_in; (void)out_size; (void)ws_size;
  const float* x   = (const float*)d_in[0];
  const float* Wq  = (const float*)d_in[1];
  const float* Wk  = (const float*)d_in[2];
  const float* Wv  = (const float*)d_in[3];
  const float* Wp  = (const float*)d_in[4];
  const float* bp  = (const float*)d_in[5];
  const float* g1  = (const float*)d_in[6];
  const float* b1  = (const float*)d_in[7];
  const float* W1  = (const float*)d_in[8];
  const float* bf1 = (const float*)d_in[9];
  const float* W2  = (const float*)d_in[10];
  const float* bf2 = (const float*)d_in[11];
  const float* g2  = (const float*)d_in[12];
  const float* b2  = (const float*)d_in[13];

  char* ws = (char*)d_ws;
  size_t off = 0;
  auto take = [&](size_t bytes) -> char* {
    char* p = ws + off;
    off += (bytes + 255) & ~(size_t)255;
    return p;
  };

  _Float16* xh  = (_Float16*)take((size_t)B_ * T_ * C_ * 2);
  _Float16* wqT = (_Float16*)take((size_t)H_ * HS_ * C_ * 2);
  _Float16* wkT = (_Float16*)take((size_t)H_ * HS_ * C_ * 2);
  _Float16* wvT = (_Float16*)take((size_t)H_ * HS_ * C_ * 2);
  _Float16* wpT = (_Float16*)take((size_t)(H_ * HS_) * C_ * 2);
  _Float16* w1T = (_Float16*)take((size_t)FF_ * C_ * 2);
  _Float16* w2T = (_Float16*)take((size_t)C_ * FF_ * 2);
  _Float16* qb  = (_Float16*)take((size_t)B_ * H_ * T_ * HS_ * 2);
  _Float16* kb  = (_Float16*)take((size_t)B_ * H_ * T_ * HS_ * 2);
  _Float16* vTb = (_Float16*)take((size_t)B_ * H_ * HS_ * T_ * 2);
  _Float16* ob  = (_Float16*)take((size_t)B_ * T_ * H_ * HS_ * 2);

  conv_x_kernel<<<(B_ * T_ * C_ + 255) / 256, 256, 0, stream>>>(x, xh);
  conv_w_kernel<<<(131072 + 255) / 256, 256, 0, stream>>>(
      Wq, Wk, Wv, Wp, W1, W2, wqT, wkT, wvT, wpT, w1T, w2T);
  qkv_kernel<<<B_ * H_ * (T_ / 16), 32, 0, stream>>>(xh, wqT, wkT, wvT, qb, kb, vTb);
  attn_kernel<<<B_ * H_ * (T_ / 16), 32, 0, stream>>>(qb, kb, vTb, ob);
  mlp_kernel<<<(B_ * T_) / 16, 32, 0, stream>>>(
      ob, wpT, x, bp, g1, b1, w1T, bf1, w2T, bf2, g2, b2, (float*)d_out);
}
